// CrossAttentionTransformer_79242146611455
// MI455X (gfx1250) — matrix operand order repaired and checked by reading, not compiled, not run on hardware
//
#include <hip/hip_runtime.h>
#include <hip/hip_bf16.h>

// ---------------------------------------------------------------------------
// CDNA5 (gfx1250, wave32) implementation of the node/triangle cross-attention
// transformer layer. All dense linears run through v_wmma_f32_16x16x32_bf16
// with fp32 accumulation; A-panels staged via LDS; weights pre-transposed to
// bf16 [N,K] once per launch (deterministic).
// ---------------------------------------------------------------------------

typedef __bf16 bf16;
typedef __attribute__((ext_vector_type(16))) __bf16 v16bf;
typedef __attribute__((ext_vector_type(8)))  float  v8f;

struct AB32 { uint4 lo, hi; };   // 32 bytes -> bit_cast to v16bf

#define DMODEL 256

// ------------------------- weight transpose + bf16 -------------------------
// wt[n*K + k] = (bf16) w[k*N + n]   (K, N are powers of two here)
__global__ __launch_bounds__(256) void k_wt(const float* __restrict__ w,
                                            bf16* __restrict__ wt,
                                            int K, int N, int kshift) {
  int i = blockIdx.x * 256 + threadIdx.x;
  if (i >= K * N) return;
  int n = i >> kshift;
  int k = i & (K - 1);
  wt[i] = (bf16)w[(size_t)k * N + n];
}

// ------------------------------- layernorm ---------------------------------
// one wave per row of 256; writes bf16 (GEMM input) and optional fp32 copy
__global__ __launch_bounds__(256) void k_ln(const float* __restrict__ x,
                                            const float* __restrict__ g,
                                            const float* __restrict__ b,
                                            bf16* __restrict__ yb,
                                            float* __restrict__ yf,
                                            int rows) {
  int lane = threadIdx.x & 31, wv = threadIdx.x >> 5;
  int row = blockIdx.x * 8 + wv;
  if (row >= rows) return;
  const float* xr = x + (size_t)row * DMODEL + lane * 8;
  float4 x0 = *(const float4*)xr, x1 = *(const float4*)(xr + 4);
  float xv[8] = {x0.x, x0.y, x0.z, x0.w, x1.x, x1.y, x1.z, x1.w};
  float s = 0.f, ss = 0.f;
#pragma unroll
  for (int j = 0; j < 8; ++j) { s += xv[j]; ss += xv[j] * xv[j]; }
  for (int o = 16; o > 0; o >>= 1) {
    s  += __shfl_xor(s,  o, 32);
    ss += __shfl_xor(ss, o, 32);
  }
  float mean = s * (1.f / 256.f);
  float var  = ss * (1.f / 256.f) - mean * mean;
  float r = rsqrtf(var + 1e-5f);
  const float* gr = g + lane * 8;
  const float* br = b + lane * 8;
  float4 g0 = *(const float4*)gr, g1 = *(const float4*)(gr + 4);
  float4 b0 = *(const float4*)br, b1 = *(const float4*)(br + 4);
  float gv[8] = {g0.x, g0.y, g0.z, g0.w, g1.x, g1.y, g1.z, g1.w};
  float bv[8] = {b0.x, b0.y, b0.z, b0.w, b1.x, b1.y, b1.z, b1.w};
  size_t o0 = (size_t)row * DMODEL + lane * 8;
#pragma unroll
  for (int j = 0; j < 8; ++j) {
    float yv = (xv[j] - mean) * r * gv[j] + bv[j];
    yb[o0 + j] = (bf16)yv;
    if (yf) yf[o0 + j] = yv;
  }
}

// --------------------------------- GEMM ------------------------------------
// C[M,N] = A[M,K]*W[K,N] + bias, A bf16 row-major, W pre-transposed bf16 [N,K].
// act==0: write fp32 C. act==1: write bf16 gelu(C) (exact erf gelu).
// Block: 256 threads = 8 waves. Block tile: 128 rows x N. Wave tile: 16 x 16.
__global__ __launch_bounds__(256) void k_gemm(const bf16* __restrict__ A,
                                              const bf16* __restrict__ Wt,
                                              const float* __restrict__ bias,
                                              float* __restrict__ Cf,
                                              bf16* __restrict__ Cb,
                                              int M, int N, int K, int act) {
  extern __shared__ bf16 lds[];
  const int ldk = K + 16;                       // pad to break DS bank conflicts
  const int tid = threadIdx.x;
  const int mbase = blockIdx.x * 128;

  // cooperative 16B-chunk copy of A[mbase:+128, 0:K] -> LDS (zero-pad tail rows)
  const int cpr = K >> 3;                       // 8 bf16 per chunk
  const int cshift = 31 - __clz(cpr);
  const int total = 128 * cpr;
  for (int i = tid; i < total; i += 256) {
    int row = i >> cshift;
    int ch  = i & (cpr - 1);
    uint4 val = make_uint4(0u, 0u, 0u, 0u);
    int gr = mbase + row;
    if (gr < M) val = *(const uint4*)(A + (size_t)gr * K + (ch << 3));
    *(uint4*)(lds + row * ldk + (ch << 3)) = val;
  }
  __syncthreads();

  const int wave = tid >> 5;
  const int lane = tid & 31;
  const int half = lane >> 4;                   // 0: lanes 0-15, 1: lanes 16-31
  const int l16  = lane & 15;

  // A fragment source: row = wave*16 + l16; K chunks at k0+half*8 and k0+16+half*8
  const bf16* arow = lds + (wave * 16 + l16) * ldk + half * 8;
  const int nTiles = N >> 4;
  const int kSteps = K >> 5;

  for (int nt = 0; nt < nTiles; ++nt) {
    const int cbase = nt << 4;
    // B fragment source: Wt row (column of W) = cbase+l16; K chunks at k0+half*8 and k0+16+half*8
    const bf16* brow = Wt + (size_t)(cbase + l16) * K + half * 8;
    float bv = bias[cbase + l16];
    v8f acc = {bv, bv, bv, bv, bv, bv, bv, bv};
    for (int kk = 0; kk < kSteps; ++kk) {
      const int k0 = kk << 5;
      AB32 au, bu;
      au.lo = *(const uint4*)(arow + k0);       // K = k0+8h   .. k0+8h+7
      au.hi = *(const uint4*)(arow + k0 + 16);  // K = k0+16+8h.. +7
      bu.lo = *(const uint4*)(brow + k0);       // K = k0+8h   .. k0+8h+7
      bu.hi = *(const uint4*)(brow + k0 + 16);  // K = k0+16+8h.. +7
      v16bf av = __builtin_bit_cast(v16bf, au);
      v16bf bvv = __builtin_bit_cast(v16bf, bu);
      acc = __builtin_amdgcn_wmma_f32_16x16x32_bf16(
          false, av, false, bvv, (short)0, acc, false, false);
    }
    // C layout: VGPR j holds row j (lanes 0-15) / row j+8 (lanes 16-31), col = l16
    const int rbase = mbase + wave * 16 + half * 8;
    const int col = cbase + l16;
#pragma unroll
    for (int j = 0; j < 8; ++j) {
      int r = rbase + j;
      if (r < M) {
        float xv = acc[j];
        if (act == 0) {
          Cf[(size_t)r * N + col] = xv;
        } else {
          float gx = 0.5f * xv * (1.0f + erff(xv * 0.70710678118654752f));
          Cb[(size_t)r * N + col] = (bf16)gx;
        }
      }
    }
  }
}

// ------------------------- edge attention kernels --------------------------
__global__ __launch_bounds__(256) void k_fill_u32(unsigned* p, unsigned v, int n) {
  int i = blockIdx.x * 256 + threadIdx.x;
  if (i < n) p[i] = v;
}

// wave per edge: logit = scale * dot(q[iq], k[ik]); float atomic max into m[dst]
__global__ __launch_bounds__(256) void k_edge_logits(const float* __restrict__ q,
                                                     const float* __restrict__ k,
                                                     const int* __restrict__ iq,
                                                     const int* __restrict__ ik,
                                                     const int* __restrict__ dst,
                                                     float* __restrict__ logit,
                                                     float* __restrict__ mbuf,
                                                     int E, float scale) {
  int lane = threadIdx.x & 31, wv = threadIdx.x >> 5;
  int e = blockIdx.x * 8 + wv;
  if (e >= E) return;
  const float* qr = q + (size_t)iq[e] * DMODEL + lane * 8;
  const float* kr = k + (size_t)ik[e] * DMODEL + lane * 8;
  float4 q0 = *(const float4*)qr, q1 = *(const float4*)(qr + 4);
  float4 k0 = *(const float4*)kr, k1 = *(const float4*)(kr + 4);
  float d = q0.x * k0.x + q0.y * k0.y + q0.z * k0.z + q0.w * k0.w +
            q1.x * k1.x + q1.y * k1.y + q1.z * k1.z + q1.w * k1.w;
  for (int o = 16; o > 0; o >>= 1) d += __shfl_xor(d, o, 32);
  if (lane == 0) {
    float l = d * scale;
    logit[e] = l;
    int di = dst[e];
    // float max via monotone int/uint bit tricks (init bits = 0xFF800000 = -inf)
    if (l >= 0.f) atomicMax((int*)(mbuf + di), __float_as_int(l));
    else          atomicMin((unsigned int*)(mbuf + di), __float_as_uint(l));
  }
}

__global__ __launch_bounds__(256) void k_edge_exp(const float* __restrict__ logit,
                                                  const float* __restrict__ mbuf,
                                                  const int* __restrict__ dst,
                                                  float* __restrict__ ew,
                                                  float* __restrict__ sbuf, int E) {
  int e = blockIdx.x * 256 + threadIdx.x;
  if (e >= E) return;
  int d = dst[e];
  float m = mbuf[d];
  if (!isfinite(m)) m = 0.f;                 // empty-segment guard (parity w/ ref)
  float v = expf(logit[e] - m);
  ew[e] = v;
  atomicAdd(sbuf + d, v);
}

// wave per edge: aggr[dst] += (ew/(s+eps)) * v[iv]
__global__ __launch_bounds__(256) void k_edge_aggr(const float* __restrict__ ew,
                                                   const float* __restrict__ sbuf,
                                                   const float* __restrict__ v,
                                                   const int* __restrict__ iv,
                                                   const int* __restrict__ dst,
                                                   float* __restrict__ aggr, int E) {
  int lane = threadIdx.x & 31, wv = threadIdx.x >> 5;
  int e = blockIdx.x * 8 + wv;
  if (e >= E) return;
  int d = dst[e];
  float a = ew[e] / (sbuf[d] + 1e-16f);
  const float* vr = v + (size_t)iv[e] * DMODEL + lane * 8;
  float* ar = aggr + (size_t)d * DMODEL + lane * 8;
  float4 v0 = *(const float4*)vr, v1 = *(const float4*)(vr + 4);
  atomicAdd(ar + 0, a * v0.x); atomicAdd(ar + 1, a * v0.y);
  atomicAdd(ar + 2, a * v0.z); atomicAdd(ar + 3, a * v0.w);
  atomicAdd(ar + 4, a * v1.x); atomicAdd(ar + 5, a * v1.y);
  atomicAdd(ar + 6, a * v1.z); atomicAdd(ar + 7, a * v1.w);
}

// ------------------------------ elementwise --------------------------------
__global__ __launch_bounds__(256) void k_add2(float* o, const float* a,
                                              const float* b, size_t n) {
  size_t i = (size_t)blockIdx.x * 256 + threadIdx.x;
  if (i < n) o[i] = a[i] + b[i];
}
__global__ __launch_bounds__(256) void k_add3(float* o, const float* a,
                                              const float* b, const float* c,
                                              size_t n) {
  size_t i = (size_t)blockIdx.x * 256 + threadIdx.x;
  if (i < n) o[i] = a[i] + b[i] + c[i];
}
__global__ __launch_bounds__(256) void k_f2b(const float* x, bf16* y, size_t n) {
  size_t i = (size_t)blockIdx.x * 256 + threadIdx.x;
  if (i < n) y[i] = (bf16)x[i];
}

// ---------------------------------------------------------------------------
// Host orchestration.
//
// d_in layout assumption (JAX pytree flatten = sorted dict keys, params
// expanded in place of the 'params' entry of setup_inputs()):
//   0 node, 1 triangle,
//   2..13  ln_{nc,nm,ns,tc,tm,ts}_{b,g}  (b before g, keys sorted)
//   14..17 mlp_n{b1,b2,w1,w2}   18..21 mlp_t{b1,b2,w1,w2}
//   22..29 n2t{bk,bo,bq,bv,wk,wo,wq,wv}
//   30..35 nsa{bk,bq,bv,wk,wq,wv}
//   36..43 t2n{bk,bo,bq,bv,wk,wo,wq,wv}
//   44..49 tsa{bk,bq,bv,wk,wq,wv}
//   50 node_edge_index, 51 tri_edge_index, 52 tn_edge_index, 53 nt_edge_index
// ---------------------------------------------------------------------------
extern "C" void kernel_launch(void* const* d_in, const int* in_sizes, int n_in,
                              void* d_out, int out_size, void* d_ws, size_t ws_size,
                              hipStream_t stream) {
  const int Dm = DMODEL;
  const float* node0 = (const float*)d_in[0];
  const float* tri0  = (const float*)d_in[1];
  const int NN = in_sizes[0] / Dm;
  const int NT = in_sizes[1] / Dm;

  const float* P[54];
  for (int i = 0; i < 54 && i < n_in; ++i) P[i] = (const float*)d_in[i];

  const int* ne = (const int*)d_in[50]; const int E_ne = in_sizes[50] / 2;
  const int* te = (const int*)d_in[51]; const int E_te = in_sizes[51] / 2;
  const int* tn = (const int*)d_in[52]; const int E_tn = in_sizes[52] / 2;
  const int* nt = (const int*)d_in[53]; const int E_nt = in_sizes[53] / 2;
  int Emax = E_ne; if (E_te > Emax) Emax = E_te; if (E_tn > Emax) Emax = E_tn;
  if (E_nt > Emax) Emax = E_nt;

  float* out_node = (float*)d_out;
  float* out_tri  = (float*)d_out + (size_t)NN * Dm;

  // ---- workspace carving (256B aligned) ----
  size_t off = 0;
  auto carve = [&](size_t bytes) -> char* {
    char* p = (char*)d_ws + off;
    off += (bytes + 255) & ~(size_t)255;
    return p;
  };

  // prep order: idx in d_in, K, N  (Wt is [N,K] bf16)
  struct WSpec { int di, K, N; };
  const WSpec wsp[18] = {
    {34, 256, 256}, {33, 256, 256}, {35, 256, 256},                  // nsa q,k,v
    {42, 256, 256}, {40, 256, 256}, {43, 256, 256}, {41, 256, 256},  // t2n q,k,v,o
    {16, 256, 1024}, {17, 1024, 256},                                // mlp_n w1,w2
    {48, 256, 256}, {47, 256, 256}, {49, 256, 256},                  // tsa q,k,v
    {28, 256, 256}, {26, 256, 256}, {29, 256, 256}, {27, 256, 256},  // n2t q,k,v,o
    {20, 256, 1024}, {21, 1024, 256}                                 // mlp_t w1,w2
  };
  bf16* wt[18];
  for (int i = 0; i < 18; ++i)
    wt[i] = (bf16*)carve((size_t)wsp[i].K * wsp[i].N * sizeof(bf16));

  bf16* h_ns = (bf16*)carve((size_t)NN * Dm * sizeof(bf16));  // LN(node0, ns) - reused twice
  bf16* h_ts = (bf16*)carve((size_t)NT * Dm * sizeof(bf16));  // LN(tri0, ts)  - reused twice
  bf16* hA   = (bf16*)carve((size_t)NT * 1024 * sizeof(bf16)); // gelu hidden
  bf16* hB   = (bf16*)carve((size_t)NT * Dm * sizeof(bf16));   // generic bf16 staging
  float* qb   = (float*)carve((size_t)NT * Dm * sizeof(float));
  float* kb   = (float*)carve((size_t)NT * Dm * sizeof(float));
  float* vb   = (float*)carve((size_t)NT * Dm * sizeof(float));
  float* aggr = (float*)carve((size_t)NT * Dm * sizeof(float));
  float* tmp1 = (float*)carve((size_t)NT * Dm * sizeof(float));
  float* stateA = (float*)carve((size_t)NT * Dm * sizeof(float));
  float* lnf  = (float*)carve((size_t)NT * Dm * sizeof(float)); // fp32 LN (cross residual)
  float* mbuf = (float*)carve((size_t)NT * sizeof(float));
  float* sbuf = (float*)carve((size_t)NT * sizeof(float));
  float* logit = (float*)carve((size_t)Emax * sizeof(float));
  float* ewb   = (float*)carve((size_t)Emax * sizeof(float));
  (void)ws_size; (void)out_size;

  // ---- weight transpose+convert (weights stay L2-resident afterwards) ----
  for (int i = 0; i < 18; ++i) {
    int K = wsp[i].K, N = wsp[i].N, tot = K * N;
    int ks = (K == 1024) ? 10 : 8;
    k_wt<<<(tot + 255) / 256, 256, 0, stream>>>(P[wsp[i].di], wt[i], K, N, ks);
  }

  auto gemm = [&](const bf16* A, const bf16* W, const float* bias,
                  float* Cf, bf16* Cb, int M, int N, int K, int act) {
    size_t shmem = (size_t)128 * (K + 16) * sizeof(bf16);
    k_gemm<<<dim3((M + 127) / 128), 256, shmem, stream>>>(A, W, bias, Cf, Cb,
                                                          M, N, K, act);
  };
  auto attn = [&](const float* q, const float* k, const float* v,
                  const int* iq, const int* ik, const int* iv, const int* dstA,
                  int E, int nseg) {
    k_fill_u32<<<(nseg + 255) / 256, 256, 0, stream>>>((unsigned*)mbuf,
                                                       0xFF800000u, nseg);
    hipMemsetAsync(sbuf, 0, (size_t)nseg * sizeof(float), stream);
    hipMemsetAsync(aggr, 0, (size_t)nseg * Dm * sizeof(float), stream);
    k_edge_logits<<<(E + 7) / 8, 256, 0, stream>>>(q, k, iq, ik, dstA, logit,
                                                   mbuf, E, 0.0625f);
    k_edge_exp<<<(E + 255) / 256, 256, 0, stream>>>(logit, mbuf, dstA, ewb,
                                                    sbuf, E);
    k_edge_aggr<<<(E + 7) / 8, 256, 0, stream>>>(ewb, sbuf, v, iv, dstA, aggr, E);
  };
  auto blk = [](size_t n) { return dim3((unsigned)((n + 255) / 256)); };

  const size_t nN = (size_t)NN * Dm, nT = (size_t)NT * Dm;

  // ---- input layernorms (reused by self- and cross-attention) ----
  k_ln<<<(NN + 7) / 8, 256, 0, stream>>>(node0, P[7], P[6], h_ns, nullptr, NN);
  k_ln<<<(NT + 7) / 8, 256, 0, stream>>>(tri0, P[13], P[12], h_ts, nullptr, NT);

  // ---- node self-attention: node = node0 + attn(LN(node0)) ----
  gemm(h_ns, wt[0], P[31], qb, nullptr, NN, 256, 256, 0);
  gemm(h_ns, wt[1], P[30], kb, nullptr, NN, 256, 256, 0);
  gemm(h_ns, wt[2], P[32], vb, nullptr, NN, 256, 256, 0);
  attn(qb, kb, vb, ne + E_ne, ne, ne, ne + E_ne, E_ne, NN);  // q@dst, k@src, v@src
  k_add2<<<blk(nN), 256, 0, stream>>>(stateA, node0, aggr, nN);

  // ---- cross tri->node: q from LN(tri0,ts)@src, k/v from LN(node,nc)@dst ----
  k_ln<<<(NN + 7) / 8, 256, 0, stream>>>(stateA, P[3], P[2], hB, lnf, NN);
  gemm(h_ts, wt[3], P[38], qb, nullptr, NT, 256, 256, 0);
  gemm(hB,   wt[4], P[36], kb, nullptr, NN, 256, 256, 0);
  gemm(hB,   wt[5], P[39], vb, nullptr, NN, 256, 256, 0);
  attn(qb, kb, vb, tn, tn + E_tn, tn + E_tn, tn + E_tn, E_tn, NN);
  k_f2b<<<blk(nN), 256, 0, stream>>>(aggr, hB, nN);
  gemm(hB, wt[6], P[37], tmp1, nullptr, NN, 256, 256, 0);
  // node2 = node + (out_proj + LN(node,nc))
  k_add3<<<blk(nN), 256, 0, stream>>>(out_node, stateA, tmp1, lnf, nN);

  // ---- node MLP: node += W2 @ gelu(W1 @ LN(node2)) ----
  k_ln<<<(NN + 7) / 8, 256, 0, stream>>>(out_node, P[5], P[4], hB, nullptr, NN);
  gemm(hB, wt[7], P[14], nullptr, hA, NN, 1024, 256, 1);   // gelu -> bf16
  gemm(hA, wt[8], P[15], tmp1, nullptr, NN, 256, 1024, 0);
  k_add2<<<blk(nN), 256, 0, stream>>>(out_node, out_node, tmp1, nN);

  // ---- tri self-attention: tri = tri0 + attn(LN(tri0)) ----
  gemm(h_ts, wt[9],  P[45], qb, nullptr, NT, 256, 256, 0);
  gemm(h_ts, wt[10], P[44], kb, nullptr, NT, 256, 256, 0);
  gemm(h_ts, wt[11], P[46], vb, nullptr, NT, 256, 256, 0);
  attn(qb, kb, vb, te + E_te, te, te, te + E_te, E_te, NT);
  k_add2<<<blk(nT), 256, 0, stream>>>(stateA, tri0, aggr, nT);

  // ---- cross node->tri: q from LN(node0,ns)@src, k/v from LN(tri,tc)@dst ----
  k_ln<<<(NT + 7) / 8, 256, 0, stream>>>(stateA, P[9], P[8], hB, lnf, NT);
  gemm(h_ns, wt[12], P[24], qb, nullptr, NN, 256, 256, 0);
  gemm(hB,   wt[13], P[22], kb, nullptr, NT, 256, 256, 0);
  gemm(hB,   wt[14], P[25], vb, nullptr, NT, 256, 256, 0);
  attn(qb, kb, vb, nt, nt + E_nt, nt + E_nt, nt + E_nt, E_nt, NT);
  k_f2b<<<blk(nT), 256, 0, stream>>>(aggr, hB, nT);
  gemm(hB, wt[15], P[23], tmp1, nullptr, NT, 256, 256, 0);
  k_add3<<<blk(nT), 256, 0, stream>>>(out_tri, stateA, tmp1, lnf, nT);

  // ---- tri MLP ----
  k_ln<<<(NT + 7) / 8, 256, 0, stream>>>(out_tri, P[11], P[10], hB, nullptr, NT);
  gemm(hB, wt[16], P[18], nullptr, hA, NT, 1024, 256, 1);
  gemm(hA, wt[17], P[19], tmp1, nullptr, NT, 256, 1024, 0);
  k_add2<<<blk(nT), 256, 0, stream>>>(out_tri, out_tri, tmp1, nT);
}
